// MemoryModule_35442070126900
// MI455X (gfx1250) — matrix operand order repaired and checked by reading, not compiled, not run on hardware
//
#include <hip/hip_runtime.h>
#include <hip/hip_fp16.h>
#include <math.h>

typedef __attribute__((ext_vector_type(16))) _Float16 v16h;
typedef __attribute__((ext_vector_type(8)))  float    v8f;

#define D_DIM 256
#define S_DIM 2048
#define N_TOT 32768
#define INV_T 20.0f   // 1/temperature

// ---------------------------------------------------------------------------
// Kernel 1: normalize memory rows -> mnorm (f16, row major, K for QK^T)
//           and memT (f16, transposed raw memory, V for PV)
// ---------------------------------------------------------------------------
__global__ void prep_mem(const float* __restrict__ mem,
                         _Float16* __restrict__ mnorm,
                         _Float16* __restrict__ memT) {
    const int s = blockIdx.x;
    const int t = threadIdx.x;          // 0..255 == dim index
    float x = mem[(size_t)s * D_DIM + t];
    float v = x * x;
    #pragma unroll
    for (int m = 16; m >= 1; m >>= 1) v += __shfl_xor(v, m, 32);
    __shared__ float red[8];
    if ((t & 31) == 0) red[t >> 5] = v;
    __syncthreads();
    __shared__ float tot;
    if (t == 0) {
        float a = 0.f;
        #pragma unroll
        for (int i = 0; i < 8; ++i) a += red[i];
        tot = a;
    }
    __syncthreads();
    const float inv = 1.0f / fmaxf(sqrtf(tot), 1e-12f);
    mnorm[(size_t)s * D_DIM + t] = (_Float16)(x * inv);
    memT [(size_t)t * S_DIM + s] = (_Float16)x;
}

// ---------------------------------------------------------------------------
// Kernel 2: zero the entropy + weights_image tail of d_out (atomics target)
// ---------------------------------------------------------------------------
__global__ void zero_tail(float* __restrict__ p, int n) {
    int i = blockIdx.x * blockDim.x + threadIdx.x;
    if (i < n) p[i] = 0.0f;
}

// ---------------------------------------------------------------------------
// Kernel 3: fused attention, scores materialized in LDS (64KB per wave).
//  Prologue: q block (16x256 f32) staged once through LDS, l2-normalized,
//            1/temperature folded into the A-fragment.
//  Pass 1: QK^T wmma once -> per-lane running max; scaled scores to LDS (f16)
//  Pass 2: LDS sweep: p = exp(s - m) in place, per-lane running sums
//  Pass 3: LDS sweep: entropy + weights_image, P-fragment straight from LDS,
//          PV wmma with unnormalized P; 1/l folded into epilogue.
// ---------------------------------------------------------------------------
__launch_bounds__(128, 1)
__global__ void attn_kernel(const float* __restrict__ feat,
                            const _Float16* __restrict__ mnorm,
                            const _Float16* __restrict__ memT,
                            float* __restrict__ out_ret,
                            float* __restrict__ out_ent,
                            float* __restrict__ out_wimg) {
    const int lane = threadIdx.x & 31;
    const int wave = threadIdx.x >> 5;
    const int wgl  = blockIdx.x * 4 + wave;   // global wave id
    const int n0   = wgl * 16;                // first query row of this wave
    const int b    = n0 >> 10;                // batch (1024 rows per image)
    const int half = lane >> 4;               // 0: lanes 0-15, 1: lanes 16-31
    const int l16  = lane & 15;

    // 4 waves x 16 rows x 2048 slots (f16) = 256KB of the 320KB WGP LDS
    __shared__ __align__(32) _Float16 sbuf[4][16][S_DIM];

    // ---- stage this wave's 16 query rows through LDS (read feat ONCE) ----
    // query[n][d] = feat[b*D*1024 + d*1024 + (n&1023)]  (stride 1024 floats)
    const int  nrow = n0 + l16;
    const float* qrow = feat + (size_t)(nrow >> 10) * D_DIM * 1024 + (nrow & 1023);
    float* qbuf = (float*)&sbuf[wave][0][0];          // [16][256] f32 scratch view

    float ss = 0.f;
    #pragma unroll 8
    for (int d = half * 128; d < half * 128 + 128; ++d) {
        const float x = qrow[(size_t)d * 1024];
        ss += x * x;
        qbuf[l16 * D_DIM + d] = x;
    }
    ss += __shfl_xor(ss, 16, 32);                     // both halves of the row
    const float qs = INV_T / fmaxf(sqrtf(ss), 1e-12f); // l2 + 1/temperature folded

    asm volatile("s_wait_dscnt 0" ::: "memory");

    // A-frag (16-bit 16x32): lane<16 -> row l16, K {0..15}; lane>=16 -> K {16..31}
    v16h qfrag[8];
    #pragma unroll
    for (int k = 0; k < 8; ++k) {
        const float* qr = qbuf + l16 * D_DIM + k * 32 + half * 16;
        v16h f;
        #pragma unroll
        for (int i = 0; i < 8; ++i) f[i]     = (_Float16)(qr[i]      * qs);
        #pragma unroll
        for (int i = 0; i < 8; ++i) f[8 + i] = (_Float16)(qr[8 + i] * qs);
        qfrag[k] = f;
    }

    asm volatile("s_wait_dscnt 0" ::: "memory");

    // ---- pass 1: QK^T once; per-lane running max; scores -> LDS ----
    float mloc[8];
    #pragma unroll
    for (int v = 0; v < 8; ++v) mloc[v] = -1e30f;

    for (int s0 = 0; s0 < S_DIM; s0 += 32) {
        v8f acc0 = {}; v8f acc1 = {};
        #pragma unroll
        for (int k = 0; k < 8; ++k) {
            v16h b0 = *(const v16h*)(mnorm + (size_t)(s0 + l16)      * D_DIM + k * 32 + half * 16);
            v16h b1 = *(const v16h*)(mnorm + (size_t)(s0 + 16 + l16) * D_DIM + k * 32 + half * 16);
            acc0 = __builtin_amdgcn_wmma_f32_16x16x32_f16(false, qfrag[k], false, b0, (short)0, acc0, false, false);
            acc1 = __builtin_amdgcn_wmma_f32_16x16x32_f16(false, qfrag[k], false, b1, (short)0, acc1, false, false);
        }
        #pragma unroll
        for (int v = 0; v < 8; ++v) {
            mloc[v] = fmaxf(mloc[v], fmaxf(acc0[v], acc1[v]));
            const int r = v + half * 8;            // C layout: M = v + 8*half, N = l16
            sbuf[wave][r][s0 + l16]      = (_Float16)acc0[v];
            sbuf[wave][r][s0 + 16 + l16] = (_Float16)acc1[v];
        }
    }
    // one cross-lane reduction for the whole S range
    float mrow[8];
    #pragma unroll
    for (int v = 0; v < 8; ++v) {
        float mx = mloc[v];
        #pragma unroll
        for (int msk = 8; msk >= 1; msk >>= 1) mx = fmaxf(mx, __shfl_xor(mx, msk, 32));
        mrow[v] = mx;
    }

    asm volatile("s_wait_dscnt 0" ::: "memory");

    // ---- pass 2: p = exp(s - m) in place; per-lane running sums ----
    float lloc[8];
    #pragma unroll
    for (int v = 0; v < 8; ++v) lloc[v] = 0.f;

    for (int s0 = 0; s0 < S_DIM; s0 += 32) {
        #pragma unroll
        for (int v = 0; v < 8; ++v) {
            const int r = v + half * 8;
            const float p0 = __expf((float)sbuf[wave][r][s0 + l16]      - mrow[v]);
            const float p1 = __expf((float)sbuf[wave][r][s0 + 16 + l16] - mrow[v]);
            sbuf[wave][r][s0 + l16]      = (_Float16)p0;
            sbuf[wave][r][s0 + 16 + l16] = (_Float16)p1;
            lloc[v] += p0 + p1;
        }
    }
    float linv[8];
    #pragma unroll
    for (int v = 0; v < 8; ++v) {
        float sum = lloc[v];
        #pragma unroll
        for (int msk = 8; msk >= 1; msk >>= 1) sum += __shfl_xor(sum, msk, 32);
        linv[v] = 1.0f / sum;
    }

    asm volatile("s_wait_dscnt 0" ::: "memory");

    // ---- pass 3: entropy + weights_image + PV (unnormalized P) ----
    v8f oacc[16];
    #pragma unroll
    for (int i = 0; i < 16; ++i) oacc[i] = (v8f){};
    float ent = 0.f;

    for (int s0 = 0; s0 < S_DIM; s0 += 32) {
        float cs0 = 0.f, cs1 = 0.f;
        #pragma unroll
        for (int v = 0; v < 8; ++v) {
            const int r = v + half * 8;
            const float w0 = (float)sbuf[wave][r][s0 + l16]      * linv[v];
            const float w1 = (float)sbuf[wave][r][s0 + 16 + l16] * linv[v];
            ent -= w0 * __logf(w0 + 1e-8f) + w1 * __logf(w1 + 1e-8f);
            cs0 += w0; cs1 += w1;
        }
        cs0 += __shfl_xor(cs0, 16, 32);
        cs1 += __shfl_xor(cs1, 16, 32);
        if (half == 0) {
            atomicAdd(&out_wimg[b * S_DIM + s0 + l16],      cs0 * (1.0f / 1024.0f));
            atomicAdd(&out_wimg[b * S_DIM + s0 + 16 + l16], cs1 * (1.0f / 1024.0f));
        }

        // P A-fragment straight from LDS (unnormalized p); cross-lane read is
        // safe: per-wave DS ops complete in order (writes were in pass 2).
        v16h pf;
        const _Float16* pr = &sbuf[wave][l16][s0 + half * 16];
        #pragma unroll
        for (int i = 0; i < 8; ++i) pf[i]     = pr[i];
        #pragma unroll
        for (int i = 0; i < 8; ++i) pf[8 + i] = pr[8 + i];

        #pragma unroll
        for (int dt = 0; dt < 16; ++dt) {
            v16h vb = *(const v16h*)(memT + (size_t)(dt * 16 + l16) * S_DIM + s0 + half * 16);
            oacc[dt] = __builtin_amdgcn_wmma_f32_16x16x32_f16(false, pf, false, vb, (short)0, oacc[dt], false, false);
        }
    }

    // ---- entropy reduction ----
    #pragma unroll
    for (int msk = 16; msk >= 1; msk >>= 1) ent += __shfl_xor(ent, msk, 32);
    if (lane == 0) atomicAdd(out_ent, ent * (1.0f / (float)N_TOT));

    // ---- fold 1/l, soft shrinkage, scatter back to (B, D, H, W) ----
    #pragma unroll
    for (int dt = 0; dt < 16; ++dt) {
        #pragma unroll
        for (int v = 0; v < 8; ++v) {
            const float x = oacc[dt][v] * linv[v];   // row of oacc == v + 8*half
            const float a = fabsf(x);
            const float lam = fminf(2.0f * 0.0025f / (a + 1e-8f), 1.0f);
            float y = fmaxf(a * (1.0f - lam), 0.0f);
            y = copysignf(y, x);
            const int nn = n0 + v + half * 8;
            const int d  = dt * 16 + l16;
            out_ret[((size_t)(nn >> 10) * D_DIM + d) * 1024 + (nn & 1023)] = y;
        }
    }
}

// ---------------------------------------------------------------------------
extern "C" void kernel_launch(void* const* d_in, const int* in_sizes, int n_in,
                              void* d_out, int out_size, void* d_ws, size_t ws_size,
                              hipStream_t stream) {
    const float* feat = (const float*)d_in[0];   // (32,256,32,32) f32
    const float* mem  = (const float*)d_in[1];   // (2048,256)     f32
    float* out = (float*)d_out;                  // retrieved | entropy | weights_image

    _Float16* mnorm = (_Float16*)d_ws;
    _Float16* memT  = (_Float16*)((char*)d_ws + (size_t)S_DIM * D_DIM * sizeof(_Float16));

    const size_t ret_elems  = (size_t)N_TOT * D_DIM;          // 8388608
    float* out_ent  = out + ret_elems;                        // 1 float
    float* out_wimg = out + ret_elems + 1;                    // 32*2048 floats
    const int tail = 1 + 32 * S_DIM;

    zero_tail<<<(tail + 255) / 256, 256, 0, stream>>>(out + ret_elems, tail);
    prep_mem<<<S_DIM, 256, 0, stream>>>(mem, mnorm, memT);
    attn_kernel<<<N_TOT / 64, 128, 0, stream>>>(feat, mnorm, memT, out, out_ent, out_wimg);
}
